// SASmambaRec_30399778521710
// MI455X (gfx1250) — matrix operand order repaired and checked by reading, not compiled, not run on hardware
//
#include <hip/hip_runtime.h>
#include <hip/hip_bf16.h>

typedef __attribute__((ext_vector_type(16))) _Float16 v16h;
typedef __attribute__((ext_vector_type(8)))  _Float16 v8h;
typedef __attribute__((ext_vector_type(8)))  float    v8f;

#define B_   64
#define L_   512
#define E_   256
#define H_   8
#define NS_  32
#define DC_  4
#define DI_  512
#define DTR_ 16
#define M_   (B_ * L_)          // 32768 token rows
#define XD_  128                // x_proj output dim padded 80 -> 128 (N % 64 == 0)

union Frag16 { v16h v; v8h h[2]; };

__device__ __forceinline__ v8f wmma16(v16h a, v16h b, v8f c) {
    return __builtin_amdgcn_wmma_f32_16x16x32_f16(false, a, false, b, (short)0, c, false, false);
}

__device__ __forceinline__ float wave_sum32(float v) {
    for (int off = 16; off; off >>= 1) v += __shfl_xor(v, off, 32);
    return v;
}

__device__ __forceinline__ int wave_id_scalar() {
    return __builtin_amdgcn_readfirstlane((int)(blockIdx.x * (blockDim.x >> 5) + (threadIdx.x >> 5)));
}

// ---------------------------------------------------------------------------
// f32 -> f16 weight conversion
// ---------------------------------------------------------------------------
__global__ void cvt_f16_kernel(const float* __restrict__ src, _Float16* __restrict__ dst, int n) {
    int i = blockIdx.x * blockDim.x + threadIdx.x;
    if (i < n) dst[i] = (_Float16)src[i];
}

// f32 [Nsrc,K] -> f16 [Ndst,K] with zero row padding (Ndst >= Nsrc)
__global__ void cvt_f16_pad_kernel(const float* __restrict__ src, _Float16* __restrict__ dst,
                                   int Nsrc, int Ndst, int K) {
    int i = blockIdx.x * blockDim.x + threadIdx.x;
    if (i >= Ndst * K) return;
    int n = i / K;
    dst[i] = (n < Nsrc) ? (_Float16)src[i] : (_Float16)0.0f;
}

// ---------------------------------------------------------------------------
// Embedding: seqs = item_emb[log_seqs]*sqrt(E) + pos_emb ; also f16 copy
// ---------------------------------------------------------------------------
__global__ void embed_kernel(const int* __restrict__ ids, const float* __restrict__ item_emb,
                             const float* __restrict__ pos_emb, float* __restrict__ seqs,
                             _Float16* __restrict__ seqs_h) {
    size_t idx = (size_t)blockIdx.x * blockDim.x + threadIdx.x;
    if (idx >= (size_t)M_ * E_) return;
    size_t row = idx / E_;
    int e = (int)(idx % E_);
    int l = (int)(row % L_);
    int id = ids[row];
    float v = item_emb[(size_t)id * E_ + e] * 16.0f + pos_emb[(size_t)l * E_ + e];
    seqs[idx] = v;
    seqs_h[idx] = (_Float16)v;
}

// ---------------------------------------------------------------------------
// WMMA GEMM:  C[M,N] = act( A[M,K](f16,rowmaj) * W[N,K](f16,rowmaj)^T + bias )
// Requires N % 64 == 0, K % 32 == 0, M % 16 == 0.
// One wave: 16-row stripe x 64 cols (4 tiles); A frag reused 4x; branch-free
// inner loop with wave-uniform (scalarized) control so EXEC stays all-1s.
// act: 0 = none, 1 = relu.  C (f32) and/or Ch (f16) written if non-null.
// ---------------------------------------------------------------------------
__global__ __launch_bounds__(256)
void wmma_gemm_kernel(const _Float16* __restrict__ A, const _Float16* __restrict__ W,
                      const float* __restrict__ bias, float* __restrict__ C,
                      _Float16* __restrict__ Ch, int M, int N, int K, int act) {
    const int lane = threadIdx.x & 31;
    const int wave = wave_id_scalar();
    const int ntg = N >> 6;
    const int mt = wave / ntg;
    const int ng = wave - mt * ntg;
    if (mt * 16 >= M) return;           // scalar branch (wave is SGPR)

    const int col  = lane & 15;
    const int half = lane >> 4;
    const int kbA  = half * 16;         // A-frag K offset
    const int kbB  = half * 16;         // B-frag K offset

    v8f zero = {0.f,0.f,0.f,0.f,0.f,0.f,0.f,0.f};
    v8f acc0 = zero, acc1 = zero, acc2 = zero, acc3 = zero;

    const _Float16* ap  = A + (size_t)(mt * 16 + col) * K + kbA;
    const _Float16* wp0 = W + (size_t)(ng * 64 + col) * K + kbB;
    const _Float16* wp1 = wp0 + (size_t)16 * K;
    const _Float16* wp2 = wp0 + (size_t)32 * K;
    const _Float16* wp3 = wp0 + (size_t)48 * K;

    for (int k0 = 0; k0 < K; k0 += 32) {
        Frag16 af;
        af.h[0] = *(const v8h*)(ap + k0);
        af.h[1] = *(const v8h*)(ap + k0 + 8);
        v16h b0 = *(const v16h*)(wp0 + k0);
        v16h b1 = *(const v16h*)(wp1 + k0);
        v16h b2 = *(const v16h*)(wp2 + k0);
        v16h b3 = *(const v16h*)(wp3 + k0);
        acc0 = wmma16(af.v, b0, acc0);
        acc1 = wmma16(af.v, b1, acc1);
        acc2 = wmma16(af.v, b2, acc2);
        acc3 = wmma16(af.v, b3, acc3);
    }

    const int mrow0 = mt * 16 + half * 8;
    v8f* accs[4] = {&acc0, &acc1, &acc2, &acc3};
    for (int t = 0; t < 4; ++t) {
        int n = ng * 64 + t * 16 + col;
        float bv = bias ? bias[n] : 0.0f;
        for (int r = 0; r < 8; ++r) {
            float v = (*accs[t])[r] + bv;
            if (act == 1) v = fmaxf(v, 0.0f);
            size_t idx = (size_t)(mrow0 + r) * N + n;
            if (C)  C[idx]  = v;
            if (Ch) Ch[idx] = (_Float16)v;
        }
    }
}

// ---------------------------------------------------------------------------
// Causal depthwise conv (DC=4) + SiLU over xz's u-half -> u (f32) and u_h (f16)
// ---------------------------------------------------------------------------
__global__ void conv_silu_kernel(const float* __restrict__ xz, const float* __restrict__ conv_w,
                                 const float* __restrict__ conv_b, float* __restrict__ u,
                                 _Float16* __restrict__ u_h) {
    size_t idx = (size_t)blockIdx.x * blockDim.x + threadIdx.x;
    if (idx >= (size_t)M_ * DI_) return;
    size_t m = idx / DI_;
    int d = (int)(idx % DI_);
    int b = (int)(m / L_), l = (int)(m % L_);
    float acc = conv_b[d];
    for (int k = 0; k < DC_; ++k) {
        int l2 = l + k - (DC_ - 1);
        if (l2 >= 0)
            acc += xz[((size_t)b * L_ + l2) * (2 * DI_) + d] * conv_w[d * DC_ + k];
    }
    float s = acc / (1.0f + __expf(-acc));   // silu
    u[idx] = s;
    u_h[idx] = (_Float16)s;
}

// ---------------------------------------------------------------------------
// dt = softplus(x_dbl[:, :16] @ dt_proj_w^T + dt_proj_b)   (K=16: VALU kernel)
// ---------------------------------------------------------------------------
__global__ void dt_kernel(const float* __restrict__ x_dbl, const float* __restrict__ dt_w,
                          const float* __restrict__ dt_b, float* __restrict__ dt) {
    __shared__ float sh[DTR_];
    size_t m = blockIdx.x;
    if (threadIdx.x < DTR_) sh[threadIdx.x] = x_dbl[m * XD_ + threadIdx.x];
    __syncthreads();
    for (int d = threadIdx.x; d < DI_; d += blockDim.x) {
        float acc = dt_b[d];
        for (int r = 0; r < DTR_; ++r) acc += sh[r] * dt_w[d * DTR_ + r];
        float sp = (acc > 20.0f) ? acc : log1pf(__expf(acc));
        dt[m * DI_ + d] = sp;
    }
}

// ---------------------------------------------------------------------------
// Selective scan: one wave per (b,d); lane = state n (N=32 == wave32)
// y = (scan_y + u*Dp) * silu(z)  -> y_h (f16) for out_proj GEMM
// ---------------------------------------------------------------------------
__global__ __launch_bounds__(256)
void scan_kernel(const float* __restrict__ u, const float* __restrict__ dt,
                 const float* __restrict__ x_dbl, const float* __restrict__ xz,
                 const float* __restrict__ A_log, const float* __restrict__ Dp,
                 _Float16* __restrict__ y_h) {
    const int lane = threadIdx.x & 31;
    const int wave = wave_id_scalar();
    const int b = wave >> 9;        // / DI_
    const int d = wave & (DI_ - 1);
    const float Ad = -__expf(A_log[d * NS_ + lane]);
    const float Dd = Dp[d];
    float h = 0.0f;
    for (int l = 0; l < L_; ++l) {
        size_t row = (size_t)b * L_ + l;
        float ut  = u[row * DI_ + d];          // wave-uniform address -> scalar load
        float dtt = dt[row * DI_ + d];
        float Bn  = x_dbl[row * XD_ + DTR_ + lane];
        float Cn  = x_dbl[row * XD_ + DTR_ + NS_ + lane];
        h = __expf(dtt * Ad) * h + dtt * Bn * ut;
        float yv = wave_sum32(h * Cn);
        if (lane == 0) {
            float zv = xz[row * (2 * DI_) + DI_ + d];
            float y = (yv + ut * Dd) * (zv / (1.0f + __expf(-zv)));
            y_h[row * DI_ + d] = (_Float16)y;
        }
    }
}

// ---------------------------------------------------------------------------
// Split qkv [M,3E] -> q_h/k_h [B*H, L, 32] f16 and vT_h [B*H, 32, L] f16
// ---------------------------------------------------------------------------
__global__ void split_qkv_kernel(const float* __restrict__ qkv, _Float16* __restrict__ q_h,
                                 _Float16* __restrict__ k_h, _Float16* __restrict__ vT_h) {
    size_t idx = (size_t)blockIdx.x * blockDim.x + threadIdx.x;
    if (idx >= (size_t)M_ * 3 * E_) return;
    size_t m = idx / (3 * E_);
    int c = (int)(idx % (3 * E_));
    int b = (int)(m / L_), l = (int)(m % L_);
    int which = c / E_;
    int ce = c % E_;
    int hh = ce / 32, dd = ce % 32;
    _Float16 v = (_Float16)qkv[idx];
    size_t bh = (size_t)b * H_ + hh;
    if (which == 0)      q_h[(bh * L_ + l) * 32 + dd] = v;
    else if (which == 1) k_h[(bh * L_ + l) * 32 + dd] = v;
    else                 vT_h[(bh * 32 + dd) * L_ + l] = v;
}

// ---------------------------------------------------------------------------
// Flash attention: one wave = 16 query rows of one (b,h). Streams K/V in
// chunks of 32 keys; scores stay in registers; P re-laid via LDS into an
// A-fragment for the P*V WMMA. No mask (reference has none).
// ---------------------------------------------------------------------------
__global__ __launch_bounds__(256)
void flash_attn_kernel(const _Float16* __restrict__ q, const _Float16* __restrict__ k,
                       const _Float16* __restrict__ vT, _Float16* __restrict__ o) {
    __shared__ __align__(32) _Float16 plds[8][16 * 32];
    const int wv = threadIdx.x >> 5;
    const int lane = threadIdx.x & 31;
    const int wave = __builtin_amdgcn_readfirstlane((int)(blockIdx.x * 8 + wv));
    const int bh = wave >> 5;          // / (L_/16)
    const int qt = wave & 31;
    const int col = lane & 15;
    const int half = lane >> 4;
    const int kbA = half * 16;
    const int kbB = half * 16;
    const int mb = half * 8;

    Frag16 qa;
    const _Float16* qp = q + (((size_t)bh * L_) + qt * 16 + col) * 32 + kbA;
    qa.h[0] = *(const v8h*)qp;
    qa.h[1] = *(const v8h*)(qp + 8);

    v8f zero = {0.f,0.f,0.f,0.f,0.f,0.f,0.f,0.f};
    v8f acc0 = zero, acc1 = zero;
    float mi[8], li[8];
    for (int r = 0; r < 8; ++r) { mi[r] = -1e30f; li[r] = 0.0f; }
    const float scale = 0.17677669529663687f;   // 1/sqrt(32)

    for (int kc = 0; kc < L_ / 32; ++kc) {
        const _Float16* kp = k + (((size_t)bh * L_) + kc * 32 + col) * 32 + kbB;
        v16h kb0 = *(const v16h*)kp;
        v16h kb1 = *(const v16h*)(kp + 16 * 32);
        v8f s0 = wmma16(qa.v, kb0, zero);
        v8f s1 = wmma16(qa.v, kb1, zero);
        for (int r = 0; r < 8; ++r) {
            float a = s0[r] * scale, bsc = s1[r] * scale;
            float mx = fmaxf(a, bsc);
            for (int off = 8; off; off >>= 1) mx = fmaxf(mx, __shfl_xor(mx, off, 32));
            float mnew = fmaxf(mi[r], mx);
            float al = __expf(mi[r] - mnew);
            float p0 = __expf(a - mnew), p1 = __expf(bsc - mnew);
            float rs = p0 + p1;
            for (int off = 8; off; off >>= 1) rs += __shfl_xor(rs, off, 32);
            li[r] = li[r] * al + rs;
            mi[r] = mnew;
            acc0[r] *= al; acc1[r] *= al;
            plds[wv][(mb + r) * 32 + col]      = (_Float16)p0;
            plds[wv][(mb + r) * 32 + col + 16] = (_Float16)p1;
        }
        __syncthreads();
        Frag16 pa;
        const _Float16* pp = &plds[wv][col * 32 + kbA];
        pa.h[0] = *(const v8h*)pp;
        pa.h[1] = *(const v8h*)(pp + 8);
        __syncthreads();
        const _Float16* vp = vT + (((size_t)bh * 32) + col) * L_ + kc * 32 + kbB;
        v16h vb0 = *(const v16h*)vp;
        v16h vb1 = *(const v16h*)(vp + 16 * L_);
        acc0 = wmma16(pa.v, vb0, acc0);
        acc1 = wmma16(pa.v, vb1, acc1);
    }

    const int b = bh / H_, hh = bh & (H_ - 1);
    for (int r = 0; r < 8; ++r) {
        float inv = 1.0f / li[r];
        size_t row = (size_t)b * L_ + qt * 16 + mb + r;
        o[row * E_ + hh * 32 + col]      = (_Float16)(acc0[r] * inv);
        o[row * E_ + hh * 32 + col + 16] = (_Float16)(acc1[r] * inv);
    }
}

// ---------------------------------------------------------------------------
// LayerNorm(a + b) per row (E=256), one wave per row; in-place into `out`
// ---------------------------------------------------------------------------
__global__ __launch_bounds__(256)
void ln_residual_kernel(const float* __restrict__ a, const float* __restrict__ b,
                        const float* __restrict__ g, const float* __restrict__ be,
                        float* __restrict__ out) {
    const int lane = threadIdx.x & 31;
    const size_t row = (size_t)wave_id_scalar();
    const float* ap = a + row * E_;
    const float* bp = b + row * E_;
    float x[8], s = 0.0f;
    for (int i = 0; i < 8; ++i) {
        int e = lane + 32 * i;
        x[i] = ap[e] + bp[e];
        s += x[i];
    }
    s = wave_sum32(s);
    float mu = s * (1.0f / E_);
    float v = 0.0f;
    for (int i = 0; i < 8; ++i) { float d = x[i] - mu; v += d * d; }
    v = wave_sum32(v) * (1.0f / E_);
    float inv = rsqrtf(v + 1e-6f);
    for (int i = 0; i < 8; ++i) {
        int e = lane + 32 * i;
        out[row * E_ + e] = (x[i] - mu) * inv * g[e] + be[e];
    }
}

// ---------------------------------------------------------------------------
// Gate: w[b] = sigmoid(mean_L(seqs[b]) . mw_w + mw_b)
// ---------------------------------------------------------------------------
__global__ void gate_kernel(const float* __restrict__ seqs, const float* __restrict__ mw_w,
                            const float* __restrict__ mw_b, float* __restrict__ w) {
    __shared__ float sh[256];
    int b = blockIdx.x, e = threadIdx.x;
    float s = 0.0f;
    for (int l = 0; l < L_; ++l) s += seqs[((size_t)b * L_ + l) * E_ + e];
    sh[e] = (s * (1.0f / L_)) * mw_w[e];
    __syncthreads();
    for (int off = 128; off; off >>= 1) {
        if (e < off) sh[e] += sh[e + off];
        __syncthreads();
    }
    if (e == 0) w[b] = 1.0f / (1.0f + __expf(-(sh[0] + mw_b[0])));
}

// ---------------------------------------------------------------------------
// combined = w[b]*mamba + (1-w[b])*attn_ln ; f32 + f16 copy
// ---------------------------------------------------------------------------
__global__ void combine_kernel(const float* __restrict__ mamba, const float* __restrict__ attn,
                               const float* __restrict__ w, float* __restrict__ comb,
                               _Float16* __restrict__ comb_h) {
    size_t idx = (size_t)blockIdx.x * blockDim.x + threadIdx.x;
    if (idx >= (size_t)M_ * E_) return;
    int b = (int)(idx / ((size_t)L_ * E_));
    float wv = w[b];
    float v = wv * mamba[idx] + (1.0f - wv) * attn[idx];
    comb[idx] = v;
    comb_h[idx] = (_Float16)v;
}

// ---------------------------------------------------------------------------
// Logits: one wave per token row; dot with gathered pos/neg embeddings
// ---------------------------------------------------------------------------
__global__ __launch_bounds__(256)
void logits_kernel(const float* __restrict__ feats, const float* __restrict__ item_emb,
                   const int* __restrict__ pos, const int* __restrict__ neg,
                   float* __restrict__ out) {
    const int lane = threadIdx.x & 31;
    const size_t row = (size_t)wave_id_scalar();
    int pid = pos[row], nid = neg[row];
    float sp = 0.0f, sn = 0.0f;
    for (int i = 0; i < 8; ++i) {
        int e = lane + 32 * i;
        float f = feats[row * E_ + e];
        sp += f * item_emb[(size_t)pid * E_ + e];
        sn += f * item_emb[(size_t)nid * E_ + e];
    }
    sp = wave_sum32(sp);
    sn = wave_sum32(sn);
    if (lane == 0) {
        out[row]      = sp;
        out[M_ + row] = sn;
    }
}

// ---------------------------------------------------------------------------
extern "C" void kernel_launch(void* const* d_in, const int* in_sizes, int n_in,
                              void* d_out, int out_size, void* d_ws, size_t ws_size,
                              hipStream_t stream) {
    (void)in_sizes; (void)n_in; (void)out_size; (void)ws_size;
    const int*   log_seqs   = (const int*)d_in[1];
    const int*   pos_seqs   = (const int*)d_in[2];
    const int*   neg_seqs   = (const int*)d_in[3];
    const float* item_emb   = (const float*)d_in[4];
    const float* pos_emb    = (const float*)d_in[5];
    const float* in_proj_w  = (const float*)d_in[6];
    const float* conv_w     = (const float*)d_in[7];
    const float* conv_b     = (const float*)d_in[8];
    const float* x_proj_w   = (const float*)d_in[9];
    const float* dt_proj_w  = (const float*)d_in[10];
    const float* dt_proj_b  = (const float*)d_in[11];
    const float* A_log      = (const float*)d_in[12];
    const float* Dp         = (const float*)d_in[13];
    const float* out_proj_w = (const float*)d_in[14];
    const float* attn_in_w  = (const float*)d_in[15];
    const float* attn_in_b  = (const float*)d_in[16];
    const float* attn_out_w = (const float*)d_in[17];
    const float* attn_out_b = (const float*)d_in[18];
    const float* ln1_g = (const float*)d_in[19];
    const float* ln1_b = (const float*)d_in[20];
    const float* mw_w  = (const float*)d_in[21];
    const float* mw_b  = (const float*)d_in[22];
    const float* ff_w1 = (const float*)d_in[23];
    const float* ff_b1 = (const float*)d_in[24];
    const float* ff_w2 = (const float*)d_in[25];
    const float* ff_b2 = (const float*)d_in[26];
    const float* ln2_g = (const float*)d_in[27];
    const float* ln2_b = (const float*)d_in[28];

    char* ws = (char*)d_ws;
    size_t off = 0;
    auto alloc = [&](size_t bytes) -> void* {
        void* p = ws + off;
        off = (off + bytes + 255) & ~(size_t)255;
        return p;
    };

    // f16 weight copies
    _Float16* w_inproj  = (_Float16*)alloc((size_t)2*DI_*E_*2);
    _Float16* w_xproj   = (_Float16*)alloc((size_t)XD_*DI_*2);      // zero-padded to 128 rows
    _Float16* w_outproj = (_Float16*)alloc((size_t)E_*DI_*2);
    _Float16* w_attnin  = (_Float16*)alloc((size_t)3*E_*E_*2);
    _Float16* w_attnout = (_Float16*)alloc((size_t)E_*E_*2);
    _Float16* w_ff1     = (_Float16*)alloc((size_t)4*E_*E_*2);
    _Float16* w_ff2     = (_Float16*)alloc((size_t)E_*4*E_*2);
    float*    gate_w    = (float*)alloc(B_*4);

    // activations
    float*    seqs    = (float*)alloc((size_t)M_*E_*4);
    _Float16* seqs_h  = (_Float16*)alloc((size_t)M_*E_*2);
    float*    xz      = (float*)alloc((size_t)M_*2*DI_*4);   // reused for qkv
    float*    u       = (float*)alloc((size_t)M_*DI_*4);
    _Float16* u_h     = (_Float16*)alloc((size_t)M_*DI_*2);
    float*    x_dbl   = (float*)alloc((size_t)M_*XD_*4);     // stride 128, cols 80..127 zero
    float*    dt      = (float*)alloc((size_t)M_*DI_*4);     // reused as h1_h (f16)
    _Float16* y_h     = (_Float16*)alloc((size_t)M_*DI_*2);
    float*    mamba   = (float*)alloc((size_t)M_*E_*4);
    _Float16* q_h     = (_Float16*)alloc((size_t)B_*H_*L_*32*2);
    _Float16* k_h     = (_Float16*)alloc((size_t)B_*H_*L_*32*2);
    _Float16* vT_h    = (_Float16*)alloc((size_t)B_*H_*32*L_*2);
    _Float16* ao_h    = (_Float16*)alloc((size_t)M_*E_*2);
    float*    attn    = (float*)alloc((size_t)M_*E_*4);      // proj out, ln1 in-place
    float*    comb    = (float*)alloc((size_t)M_*E_*4);
    _Float16* comb_h  = (_Float16*)alloc((size_t)M_*E_*2);
    float*    ff      = (float*)alloc((size_t)M_*E_*4);      // ffn2 out, ln2 in-place = feats
    float*    qkv     = xz;                                   // alias (xz free after scan)
    _Float16* h1_h    = (_Float16*)dt;                        // alias (dt free after scan)

    auto cvt = [&](const float* s, _Float16* d, int n) {
        cvt_f16_kernel<<<(n + 255) / 256, 256, 0, stream>>>(s, d, n);
    };
    auto gemm = [&](const _Float16* A, const _Float16* W, const float* bias,
                    float* C, _Float16* Ch, int M, int N, int K, int act) {
        int waves = (M / 16) * (N / 64);
        wmma_gemm_kernel<<<(waves + 7) / 8, 256, 0, stream>>>(A, W, bias, C, Ch, M, N, K, act);
    };

    // 0) weight conversions
    cvt(in_proj_w,  w_inproj,  2*DI_*E_);
    cvt_f16_pad_kernel<<<(XD_*DI_ + 255)/256, 256, 0, stream>>>(x_proj_w, w_xproj,
                                                                DTR_ + 2*NS_, XD_, DI_);
    cvt(out_proj_w, w_outproj, E_*DI_);
    cvt(attn_in_w,  w_attnin,  3*E_*E_);
    cvt(attn_out_w, w_attnout, E_*E_);
    cvt(ff_w1,      w_ff1,     4*E_*E_);
    cvt(ff_w2,      w_ff2,     E_*4*E_);

    // 1) embedding
    embed_kernel<<<(M_*E_ + 255)/256, 256, 0, stream>>>(log_seqs, item_emb, pos_emb, seqs, seqs_h);

    // 2) mamba: in_proj -> conv+silu -> x_proj -> dt -> scan -> out_proj
    gemm(seqs_h, w_inproj, nullptr, xz, nullptr, M_, 2*DI_, E_, 0);
    conv_silu_kernel<<<((size_t)M_*DI_ + 255)/256, 256, 0, stream>>>(xz, conv_w, conv_b, u, u_h);
    gemm(u_h, w_xproj, nullptr, x_dbl, nullptr, M_, XD_, DI_, 0);
    dt_kernel<<<M_, 256, 0, stream>>>(x_dbl, dt_proj_w, dt_proj_b, dt);
    scan_kernel<<<(B_*DI_)/8, 256, 0, stream>>>(u, dt, x_dbl, xz, A_log, Dp, y_h);
    gemm(y_h, w_outproj, nullptr, mamba, nullptr, M_, E_, DI_, 0);

    // 3) attention (xz/dt now free -> reuse)
    gemm(seqs_h, w_attnin, attn_in_b, qkv, nullptr, M_, 3*E_, E_, 0);
    split_qkv_kernel<<<((size_t)M_*3*E_ + 255)/256, 256, 0, stream>>>(qkv, q_h, k_h, vT_h);
    flash_attn_kernel<<<(B_*H_*(L_/16))/8, 256, 0, stream>>>(q_h, k_h, vT_h, ao_h);
    gemm(ao_h, w_attnout, attn_out_b, attn, nullptr, M_, E_, E_, 0);
    ln_residual_kernel<<<M_/8, 256, 0, stream>>>(attn, seqs, ln1_g, ln1_b, attn);

    // 4) gated merge
    gate_kernel<<<B_, 256, 0, stream>>>(seqs, mw_w, mw_b, gate_w);
    combine_kernel<<<((size_t)M_*E_ + 255)/256, 256, 0, stream>>>(mamba, attn, gate_w, comb, comb_h);

    // 5) FFN + ln2
    gemm(comb_h, w_ff1, ff_b1, nullptr, h1_h, M_, 4*E_, E_, 1);
    gemm(h1_h, w_ff2, ff_b2, ff, nullptr, M_, E_, 4*E_, 0);
    ln_residual_kernel<<<M_/8, 256, 0, stream>>>(ff, comb, ln2_g, ln2_b, ff);

    // 6) logits
    logits_kernel<<<M_/8, 256, 0, stream>>>(ff, item_emb, pos_seqs, neg_seqs, (float*)d_out);
}
